// ConvspatialAttentionBlock_2534030705304
// MI455X (gfx1250) — matrix operand order repaired and checked by reading, not compiled, not run on hardware
//
#include <hip/hip_runtime.h>
#include <hip/hip_bf16.h>

// Problem constants (from reference): B=4, C=512, N=4096, CQK=64
constexpr int kB = 4;
constexpr int kC = 512;
constexpr int kN = 4096;
constexpr int kD = 64;   // CQK

typedef __attribute__((ext_vector_type(16))) __bf16 v16bf;
typedef __attribute__((ext_vector_type(8)))  __bf16 v8bf;
typedef __attribute__((ext_vector_type(8)))  float  v8f;

#define DEV static __device__ __forceinline__

DEV v8f wmma_bf16(v16bf a, v16bf b, v8f c) {
  // (neg_a, A, neg_b, B, c_mod, C, reuse_a, reuse_b)
  return __builtin_amdgcn_wmma_f32_16x16x32_bf16(false, a, false, b, (short)0, c,
                                                 false, false);
}

DEV v16bf zero16bf() {
  v16bf r;
#pragma unroll
  for (int i = 0; i < 16; ++i) r[i] = (__bf16)0.f;
  return r;
}

// A fragment (16x32 bf16, M x K): lane holds row M = lane%16.
// halves 0..15 -> K = 2*base + 0..15,  base = (lane<16 ? 0 : 8)
// (contiguous K order, not the ISA table order; the B loads use the same order)
DEV v16bf load_a_f32(const float* __restrict__ p, int base) {
  v16bf r;
#pragma unroll
  for (int i = 0; i < 8; ++i) r[i] = (__bf16)p[2 * base + i];
#pragma unroll
  for (int i = 0; i < 8; ++i) r[8 + i] = (__bf16)p[2 * base + 8 + i];
  return r;
}

DEV v16bf load_a_bf16(const __bf16* p, int base) {
  v8bf lo = *(const v8bf*)(p + 2 * base);
  v8bf hi = *(const v8bf*)(p + 2 * base + 8);
  v16bf r;
#pragma unroll
  for (int i = 0; i < 8; ++i) { r[i] = lo[i]; r[8 + i] = hi[i]; }
  return r;
}

// B fragment (32x16 bf16, K x N): lane holds col N = lane%16,
// halves i -> K = (lane<16 ? 0 : 16) + i. Caller applies the lane-half K
// offset to the pointer; here we just pull 16 consecutive bf16 (2x b128).
DEV v16bf load_b16(const __bf16* p) {
  v8bf lo = *(const v8bf*)p;
  v8bf hi = *(const v8bf*)(p + 8);
  v16bf r;
#pragma unroll
  for (int i = 0; i < 8; ++i) { r[i] = lo[i]; r[8 + i] = hi[i]; }
  return r;
}

// Async global -> LDS copy, 16 bytes per lane (ASYNCcnt-tracked).
// VDST vgpr = wave-relative LDS byte address (low 32 bits of the generic
// shared pointer, per the flat->LDS truncation rule).
DEV void async_copy_b128(unsigned int lds_addr, const void* gaddr) {
  asm volatile("global_load_async_to_lds_b128 %0, %1, off"
               :: "v"(lds_addr), "v"((unsigned long long)(uintptr_t)gaddr)
               : "memory");
}

#define WAIT_ASYNCCNT(N) asm volatile("s_wait_asynccnt " #N ::: "memory")

// ---------------------------------------------------------------------------
// Kernel 0: x[b][c][n] f32  ->  xT[b][n][c] bf16   (tiled LDS transpose)
// ---------------------------------------------------------------------------
__global__ __launch_bounds__(256) void k_transpose_cvt(
    const float* __restrict__ x, __bf16* __restrict__ xT) {
  __shared__ float tile[32][33];
  int idx = blockIdx.x;
  const int tc = idx % (kC / 32); idx /= (kC / 32);
  const int tn = idx % (kN / 32); idx /= (kN / 32);
  const int b  = idx;
  const int lx = threadIdx.x & 31;
  const int ly = threadIdx.x >> 5;   // 0..7
#pragma unroll
  for (int j = 0; j < 4; ++j) {
    const int c = tc * 32 + ly + j * 8;
    tile[ly + j * 8][lx] = x[((size_t)b * kC + c) * kN + tn * 32 + lx];
  }
  __syncthreads();
#pragma unroll
  for (int j = 0; j < 4; ++j) {
    const int n = tn * 32 + ly + j * 8;
    xT[((size_t)b * kN + n) * kC + tc * 32 + lx] = (__bf16)tile[lx][ly + j * 8];
  }
}

// ---------------------------------------------------------------------------
// Kernel 1: QKV projections via WMMA.
//   qT[b][n][d], kT[b][n][d] (bf16, position-major) ; vB[b][c][n] (bf16)
// 40 output row-tiles of 16 (4 q + 4 k + 32 v), K=512 in 16 steps of 32.
// One 16x16 tile per wave; 8 waves/block, 5 tile-groups per nTile.
// ---------------------------------------------------------------------------
__global__ __launch_bounds__(256) void k_proj_qkv(
    const __bf16* __restrict__ xT,
    const float* __restrict__ Wq, const float* __restrict__ bq,
    const float* __restrict__ Wk, const float* __restrict__ bk,
    const float* __restrict__ Wv, const float* __restrict__ bv,
    __bf16* __restrict__ qT, __bf16* __restrict__ kT,
    __bf16* __restrict__ vB) {
  int idx = blockIdx.x;
  const int tg    = idx % 5;            idx /= 5;
  const int nTile = idx % (kN / 16);    idx /= (kN / 16);
  const int b     = idx;
  const int lane  = threadIdx.x & 31;
  const int wave  = threadIdx.x >> 5;
  const int t     = tg * 8 + wave;      // 0..39
  const int col   = lane & 15;
  const int lh    = lane >> 4;          // lane-half: 0 or 1
  const int roff  = lh * 8;
  const int n     = nTile * 16 + col;

  const float* W; const float* bias; int d0;
  if (t < 4)      { W = Wq; bias = bq; d0 = t * 16; }
  else if (t < 8) { W = Wk; bias = bk; d0 = (t - 4) * 16; }
  else            { W = Wv; bias = bv; d0 = (t - 8) * 16; }

  const float*  arow = W + (size_t)(d0 + col) * kC;       // A: M = lane%16
  const __bf16* brow = xT + ((size_t)b * kN + n) * kC;    // B: N = lane%16

  v8f acc = {};
#pragma unroll 4
  for (int kk = 0; kk < kC / 32; ++kk) {
    v16bf A  = load_a_f32(arow + kk * 32, roff);
    v16bf Bm = load_b16(brow + kk * 32 + lh * 16);
    acc = wmma_bf16(A, Bm, acc);
  }

  if (t < 8) {
    // D tile rows M = roff + r are contiguous in d -> one 16B store per lane
    __bf16* outp = (t < 4 ? qT : kT) + ((size_t)b * kN + n) * kD + d0 + roff;
    v8bf o;
#pragma unroll
    for (int r = 0; r < 8; ++r) o[r] = (__bf16)(acc[r] + bias[d0 + roff + r]);
    *(v8bf*)outp = o;
  } else {
#pragma unroll
    for (int r = 0; r < 8; ++r) {
      const int c = d0 + roff + r;
      vB[((size_t)b * kC + c) * kN + n] = (__bf16)(acc[r] + bias[c]);
    }
  }
}

// ---------------------------------------------------------------------------
// Kernel 2: fused flash attention + epilogue (gamma*read + x).
// Block = 8 waves = one 16-query tile. Wave 0: S = q^T k (K=64 -> 2 WMMA per
// 16-col tile), online softmax, posts P(16x32 bf16) + row rescale to LDS.
// All waves: rescale 4 accum tiles (64 channels each) and WMMA P @ v, with v
// tiles double-buffered in LDS via async global->LDS copies (ASYNCcnt).
// ---------------------------------------------------------------------------
__global__ __launch_bounds__(256) void k_attn_flash(
    const __bf16* __restrict__ qT, const __bf16* __restrict__ kT,
    const __bf16* __restrict__ vB, const float* __restrict__ x,
    const float* __restrict__ gamma, float* __restrict__ out) {
  __shared__ __align__(64) __bf16 Plds[16][32];
  // per-wave double-buffered v tiles: [buf][wave][c_local 0..63][m_local 0..31]
  __shared__ __align__(16) __bf16 Vlds[2][8][64][32];
  __shared__ float scale_lds[16];
  __shared__ float l_lds[16];

  const int nTile = blockIdx.x & (kN / 16 - 1);
  const int b     = blockIdx.x >> 8;            // kN/16 == 256
  const int n0    = nTile * 16;
  const int lane  = threadIdx.x & 31;
  const int wave  = threadIdx.x >> 5;
  const int col   = lane & 15;
  const int lh    = lane >> 4;
  const int roff  = lh * 8;
  const int c0    = wave * 64;                  // 4 x 16 channels per wave

  // this lane's share of the per-step async stage: 8 x 16B chunks
  // chunk e = j*32+lane: c_local = e/4, m_off = (e%4)*8
  const __bf16* vwave = vB + ((size_t)b * kC + c0) * kN;

  v16bf Aq0 = zero16bf(), Aq1 = zero16bf();
  float mrow[8], lrow[8];
#pragma unroll
  for (int r = 0; r < 8; ++r) { mrow[r] = -3.0e38f; lrow[r] = 0.f; }

  if (wave == 0) {
    const __bf16* qrow = qT + ((size_t)b * kN + n0 + col) * kD;
    Aq0 = load_a_bf16(qrow, roff);        // K (=d) 0..31
    Aq1 = load_a_bf16(qrow + 32, roff);   // K (=d) 32..63
  }

  // prologue: stage v tile for ms=0 into buffer 0
#pragma unroll
  for (int j = 0; j < 8; ++j) {
    const int e = j * 32 + lane;
    const int cl = e >> 2;
    const int mo = (e & 3) * 8;
    async_copy_b128((unsigned int)(uintptr_t)&Vlds[0][wave][cl][mo],
                    vwave + (size_t)cl * kN + mo);
  }

  v8f acc[4] = {};

  for (int ms = 0; ms < kN / 32; ++ms) {
    const int m0 = ms * 32;
    const int buf = ms & 1;
    // stage next v tile into the other buffer while we compute this step
    if (ms + 1 < kN / 32) {
#pragma unroll
      for (int j = 0; j < 8; ++j) {
        const int e = j * 32 + lane;
        const int cl = e >> 2;
        const int mo = (e & 3) * 8;
        async_copy_b128((unsigned int)(uintptr_t)&Vlds[buf ^ 1][wave][cl][mo],
                        vwave + (size_t)cl * kN + (m0 + 32) + mo);
      }
    }

    if (wave == 0) {
      v8f S[2] = {};
#pragma unroll
      for (int msub = 0; msub < 2; ++msub) {
        const __bf16* kr =
            kT + ((size_t)b * kN + m0 + msub * 16 + col) * kD + lh * 16;
        S[msub] = wmma_bf16(Aq0, load_b16(kr), S[msub]);
        S[msub] = wmma_bf16(Aq1, load_b16(kr + 32), S[msub]);
      }
      // online softmax over the 32 new columns; rows in D-layout:
      // row M = r + roff, col N = lane%16 (+16 for tile 1)
      float mnew[8], sc[8];
#pragma unroll
      for (int r = 0; r < 8; ++r) {
        float v = fmaxf(S[0][r], S[1][r]);
#pragma unroll
        for (int off = 1; off < 16; off <<= 1)
          v = fmaxf(v, __shfl_xor(v, off, 32));
        const float mn = fmaxf(mrow[r], v);
        sc[r] = __expf(mrow[r] - mn);
        mnew[r] = mn;
        mrow[r] = mn;
      }
#pragma unroll
      for (int r = 0; r < 8; ++r) {
        const float p0 = __expf(S[0][r] - mnew[r]);
        const float p1 = __expf(S[1][r] - mnew[r]);
        float s = p0 + p1;
#pragma unroll
        for (int off = 1; off < 16; off <<= 1)
          s += __shfl_xor(s, off, 32);
        lrow[r] = lrow[r] * sc[r] + s;
        Plds[r + roff][col]      = (__bf16)p0;
        Plds[r + roff][16 + col] = (__bf16)p1;
      }
      if (col == 0) {
#pragma unroll
        for (int r = 0; r < 8; ++r) scale_lds[r + roff] = sc[r];
      }
    }
    __syncthreads();

    float sc8[8];
#pragma unroll
    for (int r = 0; r < 8; ++r) sc8[r] = scale_lds[r + roff];
    // A = P from LDS (row = lane%16, contiguous K halves)
    v16bf Ap = load_a_bf16(&Plds[col][0], roff);

    // drain this step's v staging; the 8 copies for the next buffer may
    // remain in flight (ASYNCcnt is per-wave, async ops complete in order)
    if (ms + 1 < kN / 32) { WAIT_ASYNCCNT(8); } else { WAIT_ASYNCCNT(0); }

#pragma unroll
    for (int ct = 0; ct < 4; ++ct) {
#pragma unroll
      for (int r = 0; r < 8; ++r) acc[ct][r] *= sc8[r];
      const __bf16* vr = &Vlds[buf][wave][ct * 16 + col][lh * 16];
      acc[ct] = wmma_bf16(Ap, load_b16(vr), acc[ct]);
    }
    __syncthreads();
  }

  if (wave == 0 && col == 0) {
#pragma unroll
    for (int r = 0; r < 8; ++r) l_lds[r + roff] = lrow[r];
  }
  __syncthreads();

  const float g = gamma[0];
  float f8[8];
#pragma unroll
  for (int r = 0; r < 8; ++r) f8[r] = g / l_lds[r + roff];
  // D rows M = n = n0 + roff + r are contiguous in memory along n
#pragma unroll
  for (int ct = 0; ct < 4; ++ct) {
    const size_t base =
        ((size_t)b * kC + c0 + ct * 16 + col) * kN + n0 + roff;
#pragma unroll
    for (int r = 0; r < 8; ++r)
      out[base + r] = acc[ct][r] * f8[r] + x[base + r];
  }
}

// ---------------------------------------------------------------------------
extern "C" void kernel_launch(void* const* d_in, const int* in_sizes, int n_in,
                              void* d_out, int out_size, void* d_ws,
                              size_t ws_size, hipStream_t stream) {
  const float* x     = (const float*)d_in[0];
  const float* Wq    = (const float*)d_in[1];
  const float* bq    = (const float*)d_in[2];
  const float* Wk    = (const float*)d_in[3];
  const float* bk    = (const float*)d_in[4];
  const float* Wv    = (const float*)d_in[5];
  const float* bv    = (const float*)d_in[6];
  const float* gamma = (const float*)d_in[7];
  float* out = (float*)d_out;

  char* ws = (char*)d_ws;
  __bf16* xT = (__bf16*)ws; ws += (size_t)kB * kN * kC * sizeof(unsigned short);
  __bf16* qT = (__bf16*)ws; ws += (size_t)kB * kN * kD * sizeof(unsigned short);
  __bf16* kTp = (__bf16*)ws; ws += (size_t)kB * kN * kD * sizeof(unsigned short);
  __bf16* vB = (__bf16*)ws;

  k_transpose_cvt<<<kB * (kN / 32) * (kC / 32), 256, 0, stream>>>(x, xT);
  k_proj_qkv<<<kB * (kN / 16) * 5, 256, 0, stream>>>(xT, Wq, bq, Wk, bk, Wv, bv,
                                                     qT, kTp, vB);
  k_attn_flash<<<kB * (kN / 16), 256, 0, stream>>>(qT, kTp, vB, x, gamma, out);
}
